// DTFU_60129542695
// MI455X (gfx1250) — compile-verified, matrix operand order checked by reading, not run on hardware
//
#include <hip/hip_runtime.h>
#include <math.h>

// ---------------------------------------------------------------------------
// Types for CDNA5 WMMA
// ---------------------------------------------------------------------------
typedef __bf16 v16bf __attribute__((ext_vector_type(16)));
typedef float  v8f   __attribute__((ext_vector_type(8)));
typedef int    v4i   __attribute__((vector_size(16)));     // b128 chunk

#define GN    4096   // graph nodes
#define TM    128    // block tile M
#define TN    128    // block tile N
#define TK    64     // K depth per staged tile (2 x bf16 16x16x32)
#define LDTK  72     // LDS row pitch in bf16 elems (144B) -> conflict-free b128

#if __has_builtin(__builtin_amdgcn_global_load_async_to_lds_b128) && \
    __has_builtin(__builtin_amdgcn_s_wait_asynccnt)
#define USE_ASYNC_LDS 1
#else
#define USE_ASYNC_LDS 0
#endif

__device__ __forceinline__ unsigned bf16bits(float f) {
  unsigned x = __float_as_uint(f);
  return (x + 0x7FFFu + ((x >> 16) & 1u)) >> 16;       // RNE
}
__device__ __forceinline__ __bf16 u2bf(unsigned short u) {
  union { unsigned short u; __bf16 b; } c; c.u = u; return c.b;
}

// 16-byte global -> LDS copy (async path when available)
__device__ __forceinline__ void stage16(const unsigned short* g, unsigned short* l) {
#if USE_ASYNC_LDS
  __builtin_amdgcn_global_load_async_to_lds_b128(
      (__attribute__((address_space(1))) v4i*)(unsigned long long)g,
      (__attribute__((address_space(3))) v4i*)(unsigned)(unsigned long long)l,
      0, 0);
#else
  *(uint4*)l = *(const uint4*)g;
#endif
}
__device__ __forceinline__ void stage_wait() {
#if USE_ASYNC_LDS
  __builtin_amdgcn_s_wait_asynccnt(0);
#endif
}

// ---------------------------------------------------------------------------
// GEMM: C[M,N] = act( A @ B^T + bias ), operands pre-converted bf16,
// both row-major with padded K stride Kp (multiple of 64, zero-filled):
//   A : [M  x Kp]   B : [>=gridX*128 x Kp] (rows beyond N are zero)
// f32 accumulate via v_wmma_f32_16x16x32_bf16. Optional bf16 copy of C
// (Cbf, leading dim ldcbf = pad64(N), zero-filled pad cols).
// 256 threads = 8 waves (2x4); wave owns 64x32 strip -> 16 WMMA / K-step.
// Staging is fully unguarded: 4 x b128 async copies per thread per operand.
// ---------------------------------------------------------------------------
__global__ __launch_bounds__(256)
void gemm_bf16_wmma(const unsigned short* __restrict__ A,
                    const unsigned short* __restrict__ B,
                    const float* __restrict__ bias,
                    float* __restrict__ C,
                    unsigned short* __restrict__ Cbf,
                    int M, int N, int Kp, int ldcbf, int doRelu)
{
  __shared__ unsigned short As[TM * LDTK];
  __shared__ unsigned short Bs[TN * LDTK];

  const int tid   = threadIdx.x;
  const int wave  = tid >> 5;
  const int lane  = tid & 31;
  const int row16 = lane & 15;
  const int khalf = (lane >> 4) << 3;      // 0 or 8 (16-bit operand K layout)
  const int waveM = wave >> 2;             // 0..1 -> 64-row band
  const int waveN = wave & 3;              // 0..3 -> 32-col band

  const int m0 = blockIdx.y * TM;
  const int n0 = blockIdx.x * TN;

  v8f acc[4][2];
#pragma unroll
  for (int mt = 0; mt < 4; mt++)
#pragma unroll
    for (int nt = 0; nt < 2; nt++)
#pragma unroll
      for (int j = 0; j < 8; j++) acc[mt][nt][j] = 0.0f;

  const int ksteps = Kp / TK;
  for (int ks = 0; ks < ksteps; ks++) {
    const int k0 = ks * TK;

    // ---- Stage A & B tiles [128 x 64] bf16, no guards, 16B chunks ----
#pragma unroll
    for (int i = 0; i < 4; i++) {
      int q = tid + i * 256;               // 1024 chunks per tile
      int r = q >> 3, c = (q & 7) << 3;
      stage16(&A[(size_t)(m0 + r) * Kp + k0 + c], &As[r * LDTK + c]);
      stage16(&B[(size_t)(n0 + r) * Kp + k0 + c], &Bs[r * LDTK + c]);
    }
    stage_wait();
    __syncthreads();

    // ---- 16 WMMA per wave per K-step (2 K-groups x 4 m x 2 n) ----
#pragma unroll
    for (int kg = 0; kg < 2; kg++) {
      const int kb = kg * 32 + khalf;
      v16bf bfr[2];
#pragma unroll
      for (int nt = 0; nt < 2; nt++) {
        const unsigned short* p = &Bs[(waveN * 32 + nt * 16 + row16) * LDTK + kb];
#pragma unroll
        for (int e = 0; e < 8; e++) { bfr[nt][e] = u2bf(p[e]); bfr[nt][8 + e] = u2bf(p[16 + e]); }
      }
#pragma unroll
      for (int mt = 0; mt < 4; mt++) {
        v16bf af;
        const unsigned short* p = &As[(waveM * 64 + mt * 16 + row16) * LDTK + kb];
#pragma unroll
        for (int e = 0; e < 8; e++) { af[e] = u2bf(p[e]); af[8 + e] = u2bf(p[16 + e]); }
#pragma unroll
        for (int nt = 0; nt < 2; nt++)
          acc[mt][nt] = __builtin_amdgcn_wmma_f32_16x16x32_bf16(
              false, af, false, bfr[nt], (short)0, acc[mt][nt], false, false);
      }
    }
    __syncthreads();
  }

  // ---- Epilogue: row = j + 8*(lane>=16), col = lane&15 ----
#pragma unroll
  for (int nt = 0; nt < 2; nt++) {
    const int col = n0 + waveN * 32 + nt * 16 + row16;
#pragma unroll
    for (int mt = 0; mt < 4; mt++) {
#pragma unroll
      for (int j = 0; j < 8; j++) {
        int row = m0 + waveM * 64 + mt * 16 + j + ((lane >> 4) << 3);
        float v = acc[mt][nt][j];
        if (bias && col < N) v += bias[col];
        if (doRelu) v = fmaxf(v, 0.0f);
        if (col < N) C[(size_t)row * N + col] = v;          // pad cols: acc==0
        if (Cbf && col < ldcbf)
          Cbf[(size_t)row * ldcbf + col] = (unsigned short)bf16bits(v);
      }
    }
  }
}

// ---------------------------------------------------------------------------
// f32 [R x C] -> bf16 [R x Cp] with zero-filled pad columns
// ---------------------------------------------------------------------------
__global__ __launch_bounds__(256)
void cvt_bf16_pad(const float* __restrict__ in, unsigned short* __restrict__ out,
                  int R, int C, int Cp)
{
  size_t idx = (size_t)blockIdx.x * 256 + threadIdx.x;
  size_t total = (size_t)R * Cp;
  if (idx >= total) return;
  int r = (int)(idx / Cp), c = (int)(idx % Cp);
  out[idx] = (c < C) ? (unsigned short)bf16bits(in[(size_t)r * C + c]) : 0;
}

// ---------------------------------------------------------------------------
// f32 [R x C] -> transposed bf16 [Cr x Rp], zero-filled pads (LDS 32x32 tiles)
// ---------------------------------------------------------------------------
__global__ __launch_bounds__(256)
void tcvt_bf16_pad(const float* __restrict__ in, unsigned short* __restrict__ out,
                   int R, int C, int Rp, int Cr)
{
  __shared__ float tile[32][33];
  const int r0 = blockIdx.x * 32;   // input row base (= output col base)
  const int c0 = blockIdx.y * 32;   // input col base (= output row base)
  const int tx = threadIdx.x, ty = threadIdx.y;   // 32 x 8
#pragma unroll
  for (int s = 0; s < 32; s += 8) {
    int r = r0 + ty + s, c = c0 + tx;
    tile[ty + s][tx] = (r < R && c < C) ? in[(size_t)r * C + c] : 0.0f;
  }
  __syncthreads();
#pragma unroll
  for (int s = 0; s < 32; s += 8) {
    int orow = c0 + ty + s, ocol = r0 + tx;
    if (orow < Cr && ocol < Rp)
      out[(size_t)orow * Rp + ocol] = (unsigned short)bf16bits(tile[tx][ty + s]);
  }
}

// ---------------------------------------------------------------------------
// Per-row top-k on S + blend with original adjacency
// ---------------------------------------------------------------------------
__global__ __launch_bounds__(256)
void topk_blend(const float* __restrict__ S, const float* __restrict__ adj,
                float* __restrict__ out, int n, int k, float sigma)
{
  __shared__ float srow[GN];
  __shared__ float rv[256];
  __shared__ int   ri[256];
  __shared__ int   selIdx[32];
  __shared__ float selVal[32];

  const int row = blockIdx.x;
  const int tid = threadIdx.x;
  for (int c = tid; c < n; c += 256) srow[c] = S[(size_t)row * n + c];
  __syncthreads();

  for (int it = 0; it < k; it++) {
    float bv = -1.0f; int bi = n;
    for (int c = tid; c < n; c += 256) {
      float v = srow[c];
      if (v > bv || (v == bv && c < bi)) { bv = v; bi = c; }
    }
    rv[tid] = bv; ri[tid] = bi; __syncthreads();
    for (int st = 128; st > 0; st >>= 1) {
      if (tid < st) {
        float ov = rv[tid + st]; int oi = ri[tid + st];
        if (ov > rv[tid] || (ov == rv[tid] && oi < ri[tid])) { rv[tid] = ov; ri[tid] = oi; }
      }
      __syncthreads();
    }
    if (tid == 0) { selIdx[it] = ri[0]; selVal[it] = rv[0]; srow[ri[0]] = -1.0f; }
    __syncthreads();
  }

  for (int c = tid; c < n; c += 256) {
    float v = (1.0f - sigma) * adj[(size_t)row * n + c];
    for (int j = 0; j < k; j++)
      if (selIdx[j] == c) v += sigma * selVal[j];
    out[(size_t)row * n + c] = v;
  }
}

// ---------------------------------------------------------------------------
__global__ __launch_bounds__(256)
void rowsum_rsqrt(const float* __restrict__ a, float* __restrict__ d, int n)
{
  __shared__ float red[256];
  const int row = blockIdx.x, tid = threadIdx.x;
  float s = 0.0f;
  for (int c = tid; c < n; c += 256) s += a[(size_t)row * n + c];
  red[tid] = s; __syncthreads();
  for (int st = 128; st > 0; st >>= 1) {
    if (tid < st) red[tid] += red[tid + st];
    __syncthreads();
  }
  if (tid == 0) d[row] = rsqrtf(red[0]);
}

// L[i,j] = d[i]*d[j]*a[j,i] written directly as bf16 (GEMM A-operand only)
__global__ __launch_bounds__(256)
void laplacian_bf16(const float* __restrict__ a, const float* __restrict__ d,
                    unsigned short* __restrict__ L, int n)
{
  __shared__ float tile[16][17];
  const int bx = blockIdx.x * 16, by = blockIdx.y * 16;
  const int tx = threadIdx.x, ty = threadIdx.y;
  tile[ty][tx] = a[(size_t)(bx + ty) * n + (by + tx)];   // a[j, i]
  __syncthreads();
  const int i = by + ty, j = bx + tx;
  L[(size_t)i * n + j] = (unsigned short)bf16bits(d[i] * d[j] * tile[tx][ty]);
}

// h = sigmoid(g)*u + (1-sigmoid(g))*t ; writes f32 h and padded bf16 h
__global__ __launch_bounds__(256)
void fuse2_kernel(const float* __restrict__ u, const float* __restrict__ t,
                  const float* __restrict__ g, float* __restrict__ hf,
                  unsigned short* __restrict__ hbf,
                  int rows, int cols, int ldp)
{
  size_t idx = (size_t)blockIdx.x * 256 + threadIdx.x;
  size_t total = (size_t)rows * ldp;
  if (idx >= total) return;
  int r = (int)(idx / ldp), c = (int)(idx % ldp);
  if (c < cols) {
    size_t src = (size_t)r * cols + c;
    float a = 1.0f / (1.0f + expf(-g[c]));
    float v = a * u[src] + (1.0f - a) * t[src];
    hf[src] = v;
    hbf[idx] = (unsigned short)bf16bits(v);
  } else {
    hbf[idx] = 0;
  }
}

// ---------------------------------------------------------------------------
// Host orchestration
// ---------------------------------------------------------------------------
static inline int pad64(int x)  { return (x + 63)  & ~63;  }
static inline int pad128(int x) { return (x + 127) & ~127; }

static void launch_gemm(const unsigned short* A, const unsigned short* B,
                        const float* bias, float* C, unsigned short* Cbf,
                        int M, int N, int Kp, int ldcbf, int relu, hipStream_t s)
{
  dim3 grid((N + TN - 1) / TN, (M + TM - 1) / TM);
  gemm_bf16_wmma<<<grid, 256, 0, s>>>(A, B, bias, C, Cbf, M, N, Kp, ldcbf, relu);
}
static void launch_cvt(const float* in, unsigned short* out, int R, int C, int Cp,
                       hipStream_t s)
{
  size_t total = (size_t)R * Cp;
  cvt_bf16_pad<<<(unsigned)((total + 255) / 256), 256, 0, s>>>(in, out, R, C, Cp);
}
static void launch_tcvt(const float* in, unsigned short* out, int R, int C,
                        int Rp, int Cr, hipStream_t s)
{
  dim3 grid((Rp + 31) / 32, (Cr + 31) / 32), blk(32, 8);
  tcvt_bf16_pad<<<grid, blk, 0, s>>>(in, out, R, C, Rp, Cr);
}

extern "C" void kernel_launch(void* const* d_in, const int* in_sizes, int n_in,
                              void* d_out, int out_size, void* d_ws, size_t ws_size,
                              hipStream_t stream)
{
  (void)in_sizes; (void)n_in; (void)out_size; (void)ws_size;

  const float* x   = (const float*)d_in[0];
  const float* adj = (const float*)d_in[1];
  const float* aeW[8]; const float* aeB[8]; const float* gW[8]; const float* fg[8];
  for (int i = 0; i < 8; i++) {
    aeW[i] = (const float*)d_in[2 + i];
    aeB[i] = (const float*)d_in[10 + i];
    gW[i]  = (const float*)d_in[18 + i];
    fg[i]  = (const float*)d_in[26 + i];
  }

  const int N = GN;
  const int dim[9] = {500, 500, 500, 2000, 10, 2000, 500, 500, 500};
  const int K = 30;
  const float MU = 0.5f;

  // Output layout: x_bar[4096,500], z[4096,10], A_pred[N,N], h[4096,500], h1[4096,10]
  float* out   = (float*)d_out;
  float* xbar  = out;
  float* zout  = xbar + (size_t)N * 500;
  float* Apred = zout + (size_t)N * 10;
  float* hout  = Apred + (size_t)N * N;
  float* h1out = hout + (size_t)N * 500;

  // Workspace carving (256B aligned)
  char* wsb = (char*)d_ws;
  size_t off = 0;
  auto carveB = [&](size_t bytes) {
    off = (off + 255) & ~(size_t)255;
    void* p = wsb + off; off += bytes; return p;
  };
  auto carveF = [&](size_t n) { return (float*)carveB(n * 4); };
  auto carveH = [&](size_t n) { return (unsigned short*)carveB(n * 2); };

  // f32 buffers
  float* e1   = carveF((size_t)N * 500);
  float* e2   = carveF((size_t)N * 500);
  float* e3   = carveF((size_t)N * 2000);
  float* dd1  = carveF((size_t)N * 2000);
  float* dd2  = carveF((size_t)N * 500);
  float* dd3  = carveF((size_t)N * 500);
  float* tbuf = carveF((size_t)N * 2000);
  float* ubuf = carveF((size_t)N * 2000);
  float* hf   = carveF((size_t)N * 2000);
  float* Smat = carveF((size_t)N * N);
  float* Abl  = carveF((size_t)N * N);
  float* dvec = carveF((size_t)N);

  // bf16 buffers (ushort)
  unsigned short* x_bf    = carveH((size_t)N * 512);
  unsigned short* adj_bf  = carveH((size_t)N * N);
  unsigned short* curadjb = carveH((size_t)N * N);
  unsigned short* hbf     = carveH((size_t)N * 2048);
  unsigned short* tT_bf   = carveH((size_t)2048 * N);
  unsigned short* wTae[8]; unsigned short* wTg[8];
  for (int i = 0; i < 8; i++) {
    size_t sz = (size_t)pad128(dim[i + 1]) * pad64(dim[i]);
    wTae[i] = carveH(sz);
    wTg[i]  = carveH(sz);
  }
  unsigned short* e1bf  = carveH((size_t)N * 512);
  unsigned short* e2bf  = carveH((size_t)N * 512);
  unsigned short* e3bf  = carveH((size_t)N * 2048);
  unsigned short* zbf   = carveH((size_t)N * 64);
  unsigned short* dd1bf = carveH((size_t)N * 2048);
  unsigned short* dd2bf = carveH((size_t)N * 512);
  unsigned short* dd3bf = carveH((size_t)N * 512);

  // --- Operand preparation (bf16, padded) ---
  launch_cvt(x,   x_bf,   N, 500, 512, stream);
  launch_cvt(adj, adj_bf, N, N,   N,   stream);
  for (int i = 0; i < 8; i++) {
    int fin = dim[i], fout = dim[i + 1];
    launch_tcvt(aeW[i], wTae[i], fin, fout, pad64(fin), pad128(fout), stream);
    launch_tcvt(gW[i],  wTg[i],  fin, fout, pad64(fin), pad128(fout), stream);
  }

  // --- Autoencoder forward (bf16-chained) ---
  launch_gemm(x_bf,  wTae[0], aeB[0], e1,   e1bf,  N,  500,  512,  512, 1, stream);
  launch_gemm(e1bf,  wTae[1], aeB[1], e2,   e2bf,  N,  500,  512,  512, 1, stream);
  launch_gemm(e2bf,  wTae[2], aeB[2], e3,   e3bf,  N, 2000,  512, 2048, 1, stream);
  launch_gemm(e3bf,  wTae[3], aeB[3], zout, zbf,   N,   10, 2048,   64, 0, stream);
  launch_gemm(zbf,   wTae[4], aeB[4], dd1,  dd1bf, N, 2000,   64, 2048, 1, stream);
  launch_gemm(dd1bf, wTae[5], aeB[5], dd2,  dd2bf, N,  500, 2048,  512, 1, stream);
  launch_gemm(dd2bf, wTae[6], aeB[6], dd3,  dd3bf, N,  500,  512,  512, 1, stream);
  launch_gemm(dd3bf, wTae[7], aeB[7], xbar, 0,     N,  500,  512,    0, 0, stream);

  const float* tras[8] = {e1, e2, e3, zout, dd1, dd2, dd3, xbar};

  // --- GNN stack ---
  const unsigned short* h_bf = x_bf;
  const unsigned short* cadj = adj_bf;
  dim3 lapGrid(N / 16, N / 16), lapBlock(16, 16);

  for (int i = 0; i < 8; i++) {
    const int fin = dim[i], fout = dim[i + 1];
    const int kpi = pad64(fin), ldo = pad64(fout);

    // t = h @ W_i ;  transpose-convert t ;  u = relu(curadj @ t)
    launch_gemm(h_bf, wTg[i], 0, tbuf, 0, N, fout, kpi, 0, 0, stream);
    launch_tcvt(tbuf, tT_bf, N, fout, N, pad128(fout), stream);
    launch_gemm(cadj, tT_bf, 0, ubuf, 0, N, fout, N, 0, 1, stream);

    // h = sigmoid(g)*u + (1-sigmoid(g))*tras[i]   (f32 + padded bf16)
    size_t total = (size_t)N * ldo;
    fuse2_kernel<<<(unsigned)((total + 255) / 256), 256, 0, stream>>>(
        ubuf, tras[i], fg[i], hf, hbf, N, fout, ldo);
    h_bf = hbf;

    // S = relu(h @ h^T)  (both operands bf16 row-major with K stride ldo)
    launch_gemm(h_bf, h_bf, 0, Smat, 0, N, N, ldo, 0, 1, stream);

    // blend with ORIGINAL adj + per-row top-K
    float* ablOut = (i == 7) ? Apred : Abl;
    topk_blend<<<N, 256, 0, stream>>>(Smat, adj, ablOut, N, K, MU);

    if (i < 7) {
      rowsum_rsqrt<<<N, 256, 0, stream>>>(Abl, dvec, N);
      laplacian_bf16<<<lapGrid, lapBlock, 0, stream>>>(Abl, dvec, curadjb, N);
      cadj = curadjb;
    }

    if (i == 3)
      (void)hipMemcpyAsync(h1out, hf, sizeof(float) * (size_t)N * fout,
                           hipMemcpyDeviceToDevice, stream);
  }

  (void)hipMemcpyAsync(hout, hf, sizeof(float) * (size_t)N * 500,
                       hipMemcpyDeviceToDevice, stream);
}
